// BiasedMultiheadAttention_51324859187437
// MI455X (gfx1250) — matrix operand order repaired and checked by reading, not compiled, not run on hardware
//
#include <hip/hip_runtime.h>
#include <hip/hip_bf16.h>

typedef __bf16 bf16_t;
typedef __attribute__((ext_vector_type(16))) __bf16 v16bf;
typedef __attribute__((ext_vector_type(8)))  __bf16 v8bf;
typedef __attribute__((ext_vector_type(8)))  float  v8f;
typedef __attribute__((ext_vector_type(4)))  __bf16 v4bf;

#define BATCH 2
#define LSEQ  2048
#define DMODEL 1024
#define NHEAD 16
#define HDIM  64
#define THREED 3072
#define LNEG  10000.0f
#define CLAMPV 20.0f
#define SPITCH 2052           // 2048 + 4 floats pad (bank stagger: 4*m mod 64 distinct)

// ---------------- WMMA helpers ----------------

__device__ __forceinline__ v8f wmma_bf16(v16bf a, v16bf b, v8f c) {
  return __builtin_amdgcn_wmma_f32_16x16x32_bf16(false, a, false, b,
                                                 (short)0, c, false, false);
}

// A fragment (16x32, row-major source, row = lane%16):
// element j -> k = kbase + (j<8 ? half*8+j : 16+half*8+(j-8))
__device__ __forceinline__ v16bf fragA_bf16(const bf16_t* __restrict__ row,
                                            int kbase, int half) {
  v8bf a = *reinterpret_cast<const v8bf*>(row + kbase + half * 8);
  v8bf b = *reinterpret_cast<const v8bf*>(row + kbase + 16 + half * 8);
  v16bf f;
#pragma unroll
  for (int j = 0; j < 8; ++j) { f[j] = a[j]; f[j + 8] = b[j]; }
  return f;
}

// B fragment (32x16): lane holds column n = lane%16; element j -> k = kbase + (j<8 ? half*8+j : 16+half*8+(j-8))
__device__ __forceinline__ v16bf fragB_bf16(const bf16_t* __restrict__ row,
                                            int kbase, int half) {
  v8bf a = *reinterpret_cast<const v8bf*>(row + kbase + half * 8);
  v8bf b = *reinterpret_cast<const v8bf*>(row + kbase + 16 + half * 8);
  v16bf f;
#pragma unroll
  for (int j = 0; j < 8; ++j) { f[j] = a[j]; f[j + 8] = b[j]; }
  return f;
}

// ---------------- Kernel 0: fp32 -> bf16 pre-conversion ----------------
__global__ __launch_bounds__(256)
void cvt_kernel(const float* __restrict__ src, bf16_t* __restrict__ dst, int n4) {
  const int stride = gridDim.x * blockDim.x;
  for (int i = blockIdx.x * blockDim.x + threadIdx.x; i < n4; i += stride) {
    const float4 v = reinterpret_cast<const float4*>(src)[i];
    v4bf o;
    o[0] = (bf16_t)v.x; o[1] = (bf16_t)v.y; o[2] = (bf16_t)v.z; o[3] = (bf16_t)v.w;
    reinterpret_cast<v4bf*>(dst)[i] = o;
  }
}

// ---------------- Kernel 1: QKV projection ----------------
// Each wave: 16x64 output tile of qkv = x @ W_in^T + b_in (4 WMMA accumulators
// sharing one A fragment). Column group (64-aligned) lies in exactly one of
// q/k/v and one head, and the 16-row tile in one batch -> uniform scatter.
__global__ __launch_bounds__(128)
void qkv_kernel(const bf16_t* __restrict__ xb, const bf16_t* __restrict__ wib,
                const float* __restrict__ b_in,
                bf16_t* __restrict__ qws, bf16_t* __restrict__ kws,
                bf16_t* __restrict__ vtws) {
  const int lane = threadIdx.x & 31;
  const int wave = threadIdx.x >> 5;
  const int tile = blockIdx.x * 4 + wave;        // 12288 wave-tiles
  const int tiles_n = THREED / 64;               // 48
  const int mt = tile / tiles_n;
  const int nt = tile % tiles_n;
  const int mo = mt * 16, no = nt * 64;
  const int half = lane >> 4;
  const int ln16 = lane & 15;

  const bf16_t* xrow = xb  + (size_t)(mo + ln16) * DMODEL;
  const bf16_t* wrow = wib + (size_t)(no + ln16) * DMODEL;   // B[k][n] = W_in[n][k]

  v8f c0 = {}, c1 = {}, c2 = {}, c3 = {};
  for (int ko = 0; ko < DMODEL; ko += 32) {
    v16bf a  = fragA_bf16(xrow, ko, half);
    v16bf b0 = fragB_bf16(wrow,               ko, half);
    v16bf b1 = fragB_bf16(wrow + 16 * DMODEL, ko, half);
    v16bf b2 = fragB_bf16(wrow + 32 * DMODEL, ko, half);
    v16bf b3 = fragB_bf16(wrow + 48 * DMODEL, ko, half);
    c0 = wmma_bf16(a, b0, c0);
    c1 = wmma_bf16(a, b1, c1);
    c2 = wmma_bf16(a, b2, c2);
    c3 = wmma_bf16(a, b3, c3);
  }

  // uniform decode of destination
  const int sec = no >> 10;              // 0:q 1:k 2:v (uniform)
  const int h   = (no & 1023) >> 6;      // head (uniform: 64-aligned group)
  const int bb  = mo / LSEQ;             // batch (uniform: 16 | LSEQ)
  const int l0  = mo % LSEQ;
  const size_t hb = (size_t)(bb * NHEAD + h);
  v8f acc[4] = {c0, c1, c2, c3};

#pragma unroll
  for (int t = 0; t < 4; ++t) {
    const int dd = t * 16 + ln16;        // head-local feature index
    const float bv = b_in[no + t * 16 + ln16];
    if (sec == 0) {
#pragma unroll
      for (int r = 0; r < 8; ++r) {
        const int l = l0 + r + half * 8;
        qws[(hb * LSEQ + l) * HDIM + dd] = (bf16_t)(acc[t][r] + bv);
      }
    } else if (sec == 1) {
#pragma unroll
      for (int r = 0; r < 8; ++r) {
        const int l = l0 + r + half * 8;
        kws[(hb * LSEQ + l) * HDIM + dd] = (bf16_t)(acc[t][r] + bv);
      }
    } else {
#pragma unroll
      for (int r = 0; r < 8; ++r) {
        const int l = l0 + r + half * 8;
        vtws[(hb * HDIM + dd) * LSEQ + l] = (bf16_t)(acc[t][r] + bv);
      }
    }
  }
}

// ---------------- Kernel 2: attention ----------------
// One block per (b, h, q-tile of 16). Full 16x2048 fp32 score panel in LDS
// (CDNA5: 320KB/WGP). Exact reference order: bias -> mask -> max -> clip ->
// softmax -> P@V.
__global__ __launch_bounds__(256)
void attn_kernel(const float* __restrict__ bias, const int* __restrict__ pad,
                 const bf16_t* __restrict__ qws, const bf16_t* __restrict__ kws,
                 const bf16_t* __restrict__ vtws, bf16_t* __restrict__ aout) {
  __shared__ float sS[16 * SPITCH];     // ~128 KB score panel
  __shared__ float sRed[16][16];
  __shared__ float sMax[16];
  __shared__ float sSum[16];
  __shared__ float sAcc[4][16][16];     // P@V accumulators (ds_add_f32)

  const int tid  = threadIdx.x;
  const int lane = tid & 31;
  const int wave = tid >> 5;
  const int half = lane >> 4;
  const int ln16 = lane & 15;

  const int qt = blockIdx.x & 127;          // L/16 = 128
  const int bh = blockIdx.x >> 7;           // b*H + h, 0..31
  const int b  = bh >> 4;
  const int hh = bh & 15;
  const int q0 = qt * 16;

  const bf16_t* Q    = qws  + ((size_t)bh * LSEQ + q0) * HDIM;
  const bf16_t* Kp   = kws  + (size_t)bh * LSEQ * HDIM;
  const bf16_t* Vt   = vtws + (size_t)bh * HDIM * LSEQ;
  const float*  Brow = bias + ((size_t)bh * LSEQ + q0) * LSEQ;

  // zero the P@V accumulators while pass A runs
  for (int i = tid; i < 4 * 16 * 16; i += 256) ((float*)sAcc)[i] = 0.f;

  // ---- Pass A: scores = QK^T/8 + bias - 10000*pad -> LDS ----
  const v16bf qa0 = fragA_bf16(Q + ln16 * HDIM, 0, half);
  const v16bf qa1 = fragA_bf16(Q + ln16 * HDIM, 32, half);
  for (int kt = wave; kt < 128; kt += 8) {
    const int col = kt * 16 + ln16;
    if (kt + 8 < 128) {   // prefetch next iteration's streams (global_prefetch_b8)
      __builtin_prefetch(Kp + (size_t)(col + 128) * HDIM, 0, 0);
      __builtin_prefetch(Brow + (size_t)(half * 8) * LSEQ + col + 128, 0, 0);
    }
    v16bf kb0 = fragB_bf16(Kp + (size_t)col * HDIM, 0, half);
    v16bf kb1 = fragB_bf16(Kp + (size_t)col * HDIM, 32, half);
    v8f c = {};
    c = wmma_bf16(qa0, kb0, c);
    c = wmma_bf16(qa1, kb1, c);
    const float padv = LNEG * (float)pad[b * LSEQ + col];
#pragma unroll
    for (int r = 0; r < 8; ++r) {
      const int m = r + half * 8;
      sS[m * SPITCH + kt * 16 + ln16] =
          c[r] * 0.125f + Brow[(size_t)m * LSEQ + kt * 16 + ln16] - padv;
    }
  }
  __syncthreads();

  // ---- Pass B: rowmax, clip, exp, rowsum (16 threads per row) ----
  {
    const int row = tid >> 4;
    const int t   = tid & 15;
    float* srow = sS + row * SPITCH;
    float mx = -3.4e38f;
    for (int i = 0; i < 128; ++i) mx = fmaxf(mx, srow[t + i * 16]);
    sRed[row][t] = mx;
    __syncthreads();
    if (t == 0) {
      float m2 = sRed[row][0];
#pragma unroll
      for (int i = 1; i < 16; ++i) m2 = fmaxf(m2, sRed[row][i]);
      sMax[row] = m2;
    }
    __syncthreads();
    const float rm = sMax[row];
    float sum = 0.f;
    for (int i = 0; i < 128; ++i) {
      float v = srow[t + i * 16] - rm;
      v = fminf(fmaxf(v, -CLAMPV), CLAMPV);
      const float e = __expf(v);
      srow[t + i * 16] = e;
      sum += e;
    }
    sRed[row][t] = sum;
    __syncthreads();
    if (t == 0) {
      float s2 = 0.f;
#pragma unroll
      for (int i = 0; i < 16; ++i) s2 += sRed[row][i];
      sSum[row] = s2;
    }
  }
  __syncthreads();

  // ---- Pass C: P @ V. Split-K 8 ways; each wave does all 4 d-slices from
  //      one shared LDS A fragment; reduce via LDS float atomics. ----
  {
    const int k0 = wave * 256;
    v8f c0 = {}, c1 = {}, c2 = {}, c3 = {};
    const float* srow = sS + ln16 * SPITCH;
    const bf16_t* vrow = Vt + (size_t)ln16 * LSEQ;
    for (int kb = k0; kb < k0 + 256; kb += 32) {
      v16bf a;
#pragma unroll
      for (int j = 0; j < 8; ++j) a[j]     = (bf16_t)srow[kb + half * 8 + j];
#pragma unroll
      for (int j = 0; j < 8; ++j) a[j + 8] = (bf16_t)srow[kb + 16 + half * 8 + j];
      v16bf vb0 = fragB_bf16(vrow,             kb, half);
      v16bf vb1 = fragB_bf16(vrow + 16 * LSEQ, kb, half);
      v16bf vb2 = fragB_bf16(vrow + 32 * LSEQ, kb, half);
      v16bf vb3 = fragB_bf16(vrow + 48 * LSEQ, kb, half);
      c0 = wmma_bf16(a, vb0, c0);
      c1 = wmma_bf16(a, vb1, c1);
      c2 = wmma_bf16(a, vb2, c2);
      c3 = wmma_bf16(a, vb3, c3);
    }
#pragma unroll
    for (int r = 0; r < 8; ++r) {
      const int m = r + half * 8;
      atomicAdd(&sAcc[0][m][ln16], c0[r]);
      atomicAdd(&sAcc[1][m][ln16], c1[r]);
      atomicAdd(&sAcc[2][m][ln16], c2[r]);
      atomicAdd(&sAcc[3][m][ln16], c3[r]);
    }
    __syncthreads();
    if (wave < 4) {
      const int dt = wave;
#pragma unroll
      for (int r = 0; r < 8; ++r) {
        const int m = r + half * 8;
        const float v = sAcc[dt][m][ln16] / sSum[m];
        const size_t tok = (size_t)b * LSEQ + q0 + m;
        aout[tok * DMODEL + hh * HDIM + dt * 16 + ln16] = (bf16_t)v;
      }
    }
  }
}

// ---------------- Kernel 3: output projection ----------------
// Each wave: 16x64 tile of out = attn @ W_out^T + b_out (fp32 result)
__global__ __launch_bounds__(128)
void proj_kernel(const bf16_t* __restrict__ attn, const bf16_t* __restrict__ wob,
                 const float* __restrict__ b_out, float* __restrict__ out) {
  const int lane = threadIdx.x & 31;
  const int wave = threadIdx.x >> 5;
  const int tile = blockIdx.x * 4 + wave;       // 4096 wave-tiles
  const int tiles_n = DMODEL / 64;              // 16
  const int mt = tile / tiles_n;
  const int nt = tile % tiles_n;
  const int mo = mt * 16, no = nt * 64;
  const int half = lane >> 4;
  const int ln16 = lane & 15;

  const bf16_t* arow = attn + (size_t)(mo + ln16) * DMODEL;
  const bf16_t* wrow = wob  + (size_t)(no + ln16) * DMODEL;

  v8f c0 = {}, c1 = {}, c2 = {}, c3 = {};
  for (int ko = 0; ko < DMODEL; ko += 32) {
    v16bf a  = fragA_bf16(arow, ko, half);
    v16bf b0 = fragB_bf16(wrow,               ko, half);
    v16bf b1 = fragB_bf16(wrow + 16 * DMODEL, ko, half);
    v16bf b2 = fragB_bf16(wrow + 32 * DMODEL, ko, half);
    v16bf b3 = fragB_bf16(wrow + 48 * DMODEL, ko, half);
    c0 = wmma_bf16(a, b0, c0);
    c1 = wmma_bf16(a, b1, c1);
    c2 = wmma_bf16(a, b2, c2);
    c3 = wmma_bf16(a, b3, c3);
  }
  v8f acc[4] = {c0, c1, c2, c3};
#pragma unroll
  for (int t = 0; t < 4; ++t) {
    const float bv = b_out[no + t * 16 + ln16];
#pragma unroll
    for (int r = 0; r < 8; ++r) {
      const int m = r + half * 8;
      out[(size_t)(mo + m) * DMODEL + no + t * 16 + ln16] = acc[t][r] + bv;
    }
  }
}

// ---------------- Host launcher ----------------
extern "C" void kernel_launch(void* const* d_in, const int* in_sizes, int n_in,
                              void* d_out, int out_size, void* d_ws, size_t ws_size,
                              hipStream_t stream) {
  const float* x     = (const float*)d_in[0];
  const int*   pad   = (const int*)  d_in[1];
  const float* bias  = (const float*)d_in[2];
  const float* W_in  = (const float*)d_in[3];
  const float* b_in  = (const float*)d_in[4];
  const float* W_out = (const float*)d_in[5];
  const float* b_out = (const float*)d_in[6];
  float* out = (float*)d_out;

  char* ws = (char*)d_ws;
  bf16_t* qws  = (bf16_t*)(ws);                               // 8 MiB
  bf16_t* kws  = (bf16_t*)(ws + (size_t)(8)  * 1024 * 1024);  // 8 MiB
  bf16_t* vtws = (bf16_t*)(ws + (size_t)(16) * 1024 * 1024);  // 8 MiB
  bf16_t* aout = (bf16_t*)(ws + (size_t)(24) * 1024 * 1024);  // 8 MiB
  bf16_t* xb   = (bf16_t*)(ws + (size_t)(32) * 1024 * 1024);  // 8 MiB
  bf16_t* wib  = (bf16_t*)(ws + (size_t)(40) * 1024 * 1024);  // 6 MiB
  bf16_t* wob  = (bf16_t*)(ws + (size_t)(46) * 1024 * 1024);  // 2 MiB

  // one-time fp32 -> bf16 conversion of GEMM operands (RNE, same as per-use cast)
  cvt_kernel<<<1024, 256, 0, stream>>>(x,     xb,  (BATCH * LSEQ * DMODEL) / 4);
  cvt_kernel<<<1024, 256, 0, stream>>>(W_in,  wib, (THREED * DMODEL) / 4);
  cvt_kernel<<<1024, 256, 0, stream>>>(W_out, wob, (DMODEL * DMODEL) / 4);

  // 4096x3072 outputs -> 12288 16x64 wave-tiles / 4 waves per block
  qkv_kernel<<<3072, 128, 0, stream>>>(xb, wib, b_in, qws, kws, vtws);
  // one block per (b,h,q-tile): 2*16*128 = 4096
  attn_kernel<<<4096, 256, 0, stream>>>(bias, pad, qws, kws, vtws, aout);
  // 4096x1024 outputs -> 4096 16x64 wave-tiles / 4 waves per block
  proj_kernel<<<1024, 128, 0, stream>>>(aout, wob, b_out, out);
}
